// StructuredGuidedAttention_48395691491479
// MI455X (gfx1250) — matrix operand order repaired and checked by reading, not compiled, not run on hardware
//
#include <hip/hip_runtime.h>
#include <hip/hip_bf16.h>
#include <math.h>

typedef __attribute__((ext_vector_type(16))) _Float16 v16h;
typedef __attribute__((ext_vector_type(8)))  float    v8f;

#define HIDDEN 128
#define ATT_DIM 64
#define HOPS 8
#define Bsz 64
#define Tt  64
#define Nnb 64
#define Ll  (Nnb * Tt)   // 4096

// Native CDNA5 v_tanh_f32 when available (TRANS32 op, CDNA5 ISA),
// libm fallback otherwise. Guarded so compilation never breaks.
__device__ __forceinline__ float fast_tanh(float x) {
#if __has_builtin(__builtin_amdgcn_tanhf)
    return __builtin_amdgcn_tanhf(x);
#elif __has_builtin(__builtin_amdgcn_tanh_f32)
    return __builtin_amdgcn_tanh_f32(x);
#else
    return tanhf(x);
#endif
}

// -------------------------------------------------------------------------
// Kernel 1: fused H = neigh*node_tiled, s1 = H@W1^T (WMMA f16->f32),
//           s1 = (s1+b1)/scale, t = tanh(s1), s2 = t@W2^T + b2,
//           store s2 transposed [B][HOPS][L] into workspace.
// grid: 64 batches * 32 L-tiles of 128 rows; 256 threads = 8 waves,
// each wave owns a 16-row x 64-col WMMA tile.
// -------------------------------------------------------------------------
__global__ __launch_bounds__(256)
void sga_s2_kernel(const float* __restrict__ node,
                   const float* __restrict__ neigh,
                   const int*   __restrict__ nnum,
                   const float* __restrict__ W1,
                   const float* __restrict__ b1,
                   const float* __restrict__ W2,
                   const float* __restrict__ b2,
                   float* __restrict__ s2out)
{
    __shared__ alignas(32) _Float16 sW1[64][144];   // W1 as f16, padded rows (288B = 9*32B)
    __shared__ float sTanh[128][65];                // tanh(s1) tile, padded
    __shared__ float sW2[8][64];
    __shared__ float sB2[8];

    const int tid  = threadIdx.x;
    const int bx   = blockIdx.x;
    const int b    = bx >> 5;
    const int tile = bx & 31;

    const int wv   = tid >> 5;
    const int lane = tid & 31;
    const int m    = lane & 15;   // matrix row / col within 16
    const int half = lane >> 4;   // K-half select

    const int lrow = tile * 128 + wv * 16 + m;   // global L index of this lane's A row
    const int trow = lrow & (Tt - 1);            // l % T
    const float* nrp = neigh + ((size_t)b * Ll + lrow) * HIDDEN;
    const float* ndp = node  + ((size_t)b * Tt + trow) * HIDDEN;

    // Kick HBM prefetch of this lane's rows before the LDS staging phase
    __builtin_prefetch(nrp, 0, 0);
    __builtin_prefetch(nrp + 64, 0, 0);
    __builtin_prefetch(ndp, 0, 0);
    __builtin_prefetch(ndp + 64, 0, 0);

    // Stage W1 (f32 -> f16), W2, b2 into LDS (uniform trip counts: EXEC stays all-1 for WMMA)
    for (int i = tid; i < 64 * 128; i += 256)
        sW1[i >> 7][i & 127] = (_Float16)W1[i];
    for (int i = tid; i < 8 * 64; i += 256)
        sW2[i >> 6][i & 63] = W2[i];
    if (tid < 8) sB2[tid] = b2[tid];
    __syncthreads();

    const float inv_scale = rsqrtf((float)nnum[0]);

    // Build A fragments (f16 16x32 layout) for all 4 K-steps, fusing neigh*node
    v16h afr[4];
    #pragma unroll
    for (int kt = 0; kt < 4; ++kt) {
        const int k0 = kt * 32 + half * 16;
        #pragma unroll
        for (int j = 0; j < 8; ++j) {
            afr[kt][j]     = (_Float16)(nrp[k0 + j]      * ndp[k0 + j]);       // K = k0..k0+7
            afr[kt][8 + j] = (_Float16)(nrp[k0 + 8 + j] * ndp[k0 + 8 + j]); // K = k0+8..k0+15
        }
    }

    const int ncol = lane & 15;                  // B column == C column for this lane
    #pragma unroll
    for (int nt = 0; nt < 4; ++nt) {
        const int a = nt * 16 + ncol;            // attention-dim index
        v8f c = {};
        #pragma unroll
        for (int kt = 0; kt < 4; ++kt) {
            // B fragment: lane n holds col n; lanes 0-15 K=0..15, lanes 16-31 K=16..31
            const v16h bfr = *(const v16h*)&sW1[a][kt * 32 + half * 16];
            c = __builtin_amdgcn_wmma_f32_16x16x32_f16(
                    false, afr[kt], false, bfr, (short)0, c, false, false);
        }
        const float bb = b1[a];
        #pragma unroll
        for (int r = 0; r < 8; ++r) {            // C: VGPR r -> row r (lanes 0-15) / r+8 (16-31)
            const float s1v = (c[r] + bb) * inv_scale;
            sTanh[wv * 16 + r + 8 * half][a] = fast_tanh(s1v);
        }
    }
    __syncthreads();

    // s2 = tanh(s1) @ W2^T + b2 ; write transposed [b][h][l].
    // Fixed hop per thread -> W2 row is loop-invariant (register-cached).
    const int h    = tid & 7;
    const int row0 = tid >> 3;                   // 0..31
    #pragma unroll
    for (int rr = 0; rr < 4; ++rr) {
        const int row = row0 + rr * 32;
        float acc = sB2[h];
        #pragma unroll 8
        for (int a2 = 0; a2 < 64; ++a2) acc += sTanh[row][a2] * sW2[h][a2];
        const int lg = tile * 128 + row;
        s2out[(((size_t)b * HOPS + h) << 12) + lg] = acc;
    }
}

// ------------------------------------------------------------------
// Kernel 2a: per (b,h) softmax stats (max, sum-of-exp) over L=4096
// ------------------------------------------------------------------
__global__ __launch_bounds__(256)
void sga_softstats_kernel(const float* __restrict__ s2, float* __restrict__ stats)
{
    __shared__ float red[256];
    const int tid = threadIdx.x;
    const int bh  = blockIdx.x;
    const float* row = s2 + ((size_t)bh << 12);

    float mx = -INFINITY;
    for (int l = tid; l < Ll; l += 256) mx = fmaxf(mx, row[l]);
    red[tid] = mx; __syncthreads();
    for (int s = 128; s > 0; s >>= 1) {
        if (tid < s) red[tid] = fmaxf(red[tid], red[tid + s]);
        __syncthreads();
    }
    mx = red[0];
    __syncthreads();

    float sum = 0.f;
    for (int l = tid; l < Ll; l += 256) sum += __expf(row[l] - mx);
    red[tid] = sum; __syncthreads();
    for (int s = 128; s > 0; s >>= 1) {
        if (tid < s) red[tid] += red[tid + s];
        __syncthreads();
    }
    if (tid == 0) { stats[bh * 2] = mx; stats[bh * 2 + 1] = red[0]; }
}

// ------------------------------------------------------------------
// Kernel 2b: A = softmax(s2) written [b][h][l]; BW[b][l] = sum_h A
// ------------------------------------------------------------------
__global__ __launch_bounds__(256)
void sga_softmax_bw_kernel(const float* __restrict__ s2, const float* __restrict__ stats,
                           float* __restrict__ Aout, float* __restrict__ BWout)
{
    const int idx = blockIdx.x * 256 + threadIdx.x;   // 0 .. B*L-1
    const int b = idx >> 12;
    const int l = idx & 4095;
    float bw = 0.f;
    #pragma unroll
    for (int h = 0; h < HOPS; ++h) {
        const int bh = b * HOPS + h;
        const float inv = 1.f / stats[bh * 2 + 1];
        const float v = __expf(s2[((size_t)bh << 12) + l] - stats[bh * 2]) * inv;
        Aout[((size_t)bh << 12) + l] = v;
        bw += v;
    }
    BWout[idx] = bw;
}

// ------------------------------------------------------------------
// Kernel 2c: per-batch ||A A^T - I||_F^2 -> penal_b[b] (deterministic)
// ------------------------------------------------------------------
__global__ __launch_bounds__(256)
void sga_penal_kernel(const float* __restrict__ Aout, float* __restrict__ penal_b)
{
    __shared__ float red[256];
    __shared__ float pairv[36];
    const int tid = threadIdx.x;
    const int b   = blockIdx.x;

    float acc[36];
    #pragma unroll
    for (int p = 0; p < 36; ++p) acc[p] = 0.f;

    const float* Ab = Aout + (((size_t)b * HOPS) << 12);
    for (int l = tid; l < Ll; l += 256) {
        float a[8];
        #pragma unroll
        for (int h = 0; h < 8; ++h) a[h] = Ab[((size_t)h << 12) + l];
        int p = 0;
        #pragma unroll
        for (int h = 0; h < 8; ++h)
            #pragma unroll
            for (int g = h; g < 8; ++g)
                acc[p++] += a[h] * a[g];
    }
    for (int p = 0; p < 36; ++p) {
        red[tid] = acc[p]; __syncthreads();
        for (int s = 128; s > 0; s >>= 1) {
            if (tid < s) red[tid] += red[tid + s];
            __syncthreads();
        }
        if (tid == 0) pairv[p] = red[0];
        __syncthreads();
    }
    if (tid == 0) {
        float pe = 0.f; int p = 0;
        for (int h = 0; h < 8; ++h)
            for (int g = h; g < 8; ++g) {
                const float v = pairv[p++] - (h == g ? 1.f : 0.f);
                pe += (h == g ? 1.f : 2.f) * v * v;   // off-diagonals counted twice
            }
        penal_b[b] = pe;
    }
}

__global__ __launch_bounds__(64)
void sga_penal_sum_kernel(const float* __restrict__ penal_b, float* __restrict__ out)
{
    __shared__ float red[64];
    const int tid = threadIdx.x;
    red[tid] = penal_b[tid];
    __syncthreads();
    for (int s = 32; s > 0; s >>= 1) {
        if (tid < s) red[tid] += red[tid + s];
        __syncthreads();
    }
    if (tid == 0) out[0] = red[0];
}

extern "C" void kernel_launch(void* const* d_in, const int* in_sizes, int n_in,
                              void* d_out, int out_size, void* d_ws, size_t ws_size,
                              hipStream_t stream)
{
    const float* node  = (const float*)d_in[0];
    const float* neigh = (const float*)d_in[1];
    const int*   nnum  = (const int*)  d_in[2];
    const float* W1    = (const float*)d_in[3];
    const float* b1    = (const float*)d_in[4];
    const float* W2    = (const float*)d_in[5];
    const float* b2    = (const float*)d_in[6];

    float* out = (float*)d_out;
    const size_t BW_off    = 0;
    const size_t A_off     = (size_t)Bsz * Ll;                       // 262144
    const size_t neigh_off = A_off + (size_t)Bsz * HOPS * Ll;        // 2359296
    const size_t penal_off = neigh_off + (size_t)Bsz * Ll * HIDDEN;  // 35913728

    float* ws      = (float*)d_ws;
    float* s2      = ws;                                   // B*HOPS*L floats (8 MB)
    float* stats   = ws + (size_t)Bsz * HOPS * Ll;         // 512*2 floats
    float* penal_b = stats + (size_t)Bsz * HOPS * 2;       // 64 floats

    // neigh output is an identity reshape of the input: D2D copy (graph-capture safe)
    hipMemcpyAsync(out + neigh_off, neigh,
                   (size_t)Bsz * Ll * HIDDEN * sizeof(float),
                   hipMemcpyDeviceToDevice, stream);

    sga_s2_kernel<<<Bsz * 32, 256, 0, stream>>>(node, neigh, nnum, W1, b1, W2, b2, s2);
    sga_softstats_kernel<<<Bsz * HOPS, 256, 0, stream>>>(s2, stats);
    sga_softmax_bw_kernel<<<(Bsz * Ll) / 256, 256, 0, stream>>>(s2, stats,
                                                                out + A_off, out + BW_off);
    sga_penal_kernel<<<Bsz, 256, 0, stream>>>(out + A_off, penal_b);
    sga_penal_sum_kernel<<<1, 64, 0, stream>>>(penal_b, out + penal_off);
}
